// Model_70471823392987
// MI455X (gfx1250) — matrix operand order repaired and checked by reading, not compiled, not run on hardware
//
#include <hip/hip_runtime.h>
#include <hip/hip_bf16.h>

typedef _Float16 v16h __attribute__((ext_vector_type(16)));
typedef _Float16 v8h  __attribute__((ext_vector_type(8)));
typedef _Float16 v4h  __attribute__((ext_vector_type(4)));
typedef float    v8f  __attribute__((ext_vector_type(8)));
typedef float    v4f  __attribute__((ext_vector_type(4)));

#define LDS_STR 72   // padded row stride in f16 elements (144 B: 16B-aligned rows, bank spread)

static __device__ __forceinline__ v16h cat8(v8h lo, v8h hi) {
    return __builtin_shufflevector(lo, hi, 0,1,2,3,4,5,6,7,8,9,10,11,12,13,14,15);
}
static __device__ __forceinline__ v8h ld8(const _Float16* p) {
    return *(const v8h*)p;
}

// One workgroup per 64x64 diagonal block. 128 threads = 4 waves (wave32).
// Wave w computes output rows [16w, 16w+16).
__global__ __launch_bounds__(128)
void lightning_diag_attn(const float* __restrict__ Q,
                         const float* __restrict__ K,
                         const float* __restrict__ V,
                         const float* __restrict__ S,
                         float* __restrict__ O,
                         int G, int H)
{
    __shared__ _Float16 lq[64 * LDS_STR];   // q tile, row-major f16
    __shared__ _Float16 lk[64 * LDS_STR];   // k tile, row-major f16
    __shared__ _Float16 lv[64 * LDS_STR];   // v tile TRANSPOSED: lv[e*STR + j]
    __shared__ _Float16 lp[64 * LDS_STR];   // decayed S (P), row-major f16

    const int tid  = threadIdx.x;
    const int lane = tid & 31;
    const int wv   = tid >> 5;
    const int hi   = lane >> 4;   // half-wave select
    const int ln   = lane & 15;

    const int blk = blockIdx.x;            // (bh*G + g)
    const int h   = (blk / G) % H;
    const size_t base = (size_t)blk * (64 * 64);   // [b,h,n,d] flattens to blk*4096

    const float c = S[h] * 1.44269504088896340736f;  // s_h * log2(e)

    // ---- Stage 1: global f32 -> LDS f16 (q,k row-major; v transposed) ----
    #pragma unroll
    for (int it = 0; it < 8; ++it) {
        const int cidx = tid + it * 128;        // float4 chunk id, 0..1023
        const int row  = cidx >> 4;
        const int col  = (cidx & 15) << 2;
        const v4f fq = *(const v4f*)(Q + base + (size_t)cidx * 4);
        const v4f fk = *(const v4f*)(K + base + (size_t)cidx * 4);
        const v4f fv = *(const v4f*)(V + base + (size_t)cidx * 4);
        v4h hq = { (_Float16)fq[0], (_Float16)fq[1], (_Float16)fq[2], (_Float16)fq[3] };
        v4h hk = { (_Float16)fk[0], (_Float16)fk[1], (_Float16)fk[2], (_Float16)fk[3] };
        *(v4h*)&lq[row * LDS_STR + col] = hq;
        *(v4h*)&lk[row * LDS_STR + col] = hk;
        lv[(col + 0) * LDS_STR + row] = (_Float16)fv[0];
        lv[(col + 1) * LDS_STR + row] = (_Float16)fv[1];
        lv[(col + 2) * LDS_STR + row] = (_Float16)fv[2];
        lv[(col + 3) * LDS_STR + row] = (_Float16)fv[3];
    }
    __syncthreads();

    const v8f zero8 = {0.f,0.f,0.f,0.f,0.f,0.f,0.f,0.f};

    // ---- GEMM1: S_strip(16x64) = q_strip(16x64) @ k^T(64x64) ----
    // A layout (16-bit, 16x32): lane<16 holds K 0..7 & 16..23; lane>=16 holds K 8..15 & 24..31
    const int arow = 16 * wv + ln;
    const _Float16* qr = &lq[arow * LDS_STR];
    const v16h a0 = cat8(ld8(qr +      hi * 8), ld8(qr + 16 + hi * 8));  // K 0..31
    const v16h a1 = cat8(ld8(qr + 32 + hi * 8), ld8(qr + 48 + hi * 8));  // K 32..63

    v8f acc[4];
    #pragma unroll
    for (int jt = 0; jt < 4; ++jt) {
        // B layout (32x16): lane n = column n; lanes<16 hold K 0..7 & 16..23, lanes>=16 hold K 8..15 & 24..31.
        // B = k^T, so column n of B is row (16jt+n) of k -> contiguous in LDS.
        const _Float16* kr = &lk[(16 * jt + ln) * LDS_STR];
        v8f a = zero8;
        const v16h b0 = cat8(ld8(kr + hi * 8), ld8(kr + 16 + hi * 8));       // K 0..31
        a = __builtin_amdgcn_wmma_f32_16x16x32_f16(false, a0, false, b0, (short)0, a, false, false);
        const v16h b1 = cat8(ld8(kr + 32 + hi * 8), ld8(kr + 48 + hi * 8));      // K 32..63
        a = __builtin_amdgcn_wmma_f32_16x16x32_f16(false, a1, false, b1, (short)0, a, false, false);
        acc[jt] = a;
    }

    // ---- decay + causal mask, write P (f16) to LDS ----
    // C layout: VGPR r -> row 16w + 8*hi + r, lane -> column 16jt + ln.
    // Factor exp2(-c*(d0+r)) = exp2(-c*d0) * exp2(-c*r): 12 unconditional
    // v_exp_f32 per wave (hoisted out of the mask), so the per-element mask
    // is a plain v_cmp + v_cndmask instead of predicated EXEC regions.
    // Intermediate exp2(-c*d0) <= 2^91 stays finite in f32.
    const int irow0 = 16 * wv + 8 * hi;
    float rfac[8];
    #pragma unroll
    for (int r = 0; r < 8; ++r)
        rfac[r] = __builtin_amdgcn_exp2f(-c * (float)r);

    #pragma unroll
    for (int jt = 0; jt < 4; ++jt) {
        const int j  = 16 * jt + ln;
        const int d0 = irow0 - j;                  // diff = d0 + r
        const float e0 = __builtin_amdgcn_exp2f(-c * (float)d0);
        #pragma unroll
        for (int r = 0; r < 8; ++r) {
            const float p = (d0 + r >= 0) ? acc[jt][r] * (e0 * rfac[r]) : 0.0f;
            lp[(irow0 + r) * LDS_STR + j] = (_Float16)p;
        }
    }
    __syncthreads();   // cheap; guarantees P stores visible before A-layout reload

    // ---- GEMM2: O_strip(16x64) = P_strip(16x64) @ v(64x64) ----
    const _Float16* pr = &lp[arow * LDS_STR];
    const v16h pa0 = cat8(ld8(pr +      hi * 8), ld8(pr + 16 + hi * 8));
    const v16h pa1 = cat8(ld8(pr + 32 + hi * 8), ld8(pr + 48 + hi * 8));

    float* op = O + base;
    #pragma unroll
    for (int et = 0; et < 4; ++et) {
        // B = v (64x16 tile): column e of B is row e of transposed LDS copy -> contiguous in K
        const _Float16* vr = &lv[(16 * et + ln) * LDS_STR];
        v8f o = zero8;
        const v16h b0 = cat8(ld8(vr + hi * 8), ld8(vr + 16 + hi * 8));       // K 0..31
        o = __builtin_amdgcn_wmma_f32_16x16x32_f16(false, pa0, false, b0, (short)0, o, false, false);
        const v16h b1 = cat8(ld8(vr + 32 + hi * 8), ld8(vr + 48 + hi * 8));      // K 32..63
        o = __builtin_amdgcn_wmma_f32_16x16x32_f16(false, pa1, false, b1, (short)0, o, false, false);
        #pragma unroll
        for (int r = 0; r < 8; ++r)
            op[(size_t)(irow0 + r) * 64 + 16 * et + ln] = o[r];
    }
}

extern "C" void kernel_launch(void* const* d_in, const int* in_sizes, int n_in,
                              void* d_out, int out_size, void* d_ws, size_t ws_size,
                              hipStream_t stream) {
    (void)n_in; (void)out_size; (void)d_ws; (void)ws_size;
    const float* q = (const float*)d_in[0];
    const float* k = (const float*)d_in[1];
    const float* v = (const float*)d_in[2];
    const float* s = (const float*)d_in[3];
    float* out = (float*)d_out;

    // Reference dims: b=2, h=16, n=8192, d=e=64, BLOCK=64
    const int H = in_sizes[3];                 // 16 heads (s is per-head)
    const int G = 8192 / 64;                   // 128 diagonal blocks per (b,h)
    const int nblocks = in_sizes[0] / (64 * 64); // b*h*G = 4096

    hipLaunchKernelGGL(lightning_diag_attn, dim3(nblocks), dim3(128), 0, stream,
                       q, k, v, s, out, G, H);
}
